// AttentionBasedInflationBlock_16286515987237
// MI455X (gfx1250) — matrix operand order repaired and checked by reading, not compiled, not run on hardware
//
#include <hip/hip_runtime.h>

// ---------------------------------------------------------------------------
// AttentionBasedInflationBlock for MI455X (gfx1250, wave32, WMMA + TDM)
//   x:[1,8,96,96,256] f32 -> out f32, via bf16 intermediates,
//   v_wmma_f32_16x16x32_bf16 for all GEMMs, tensor_load_to_lds for weights.
// ---------------------------------------------------------------------------

typedef __attribute__((ext_vector_type(16))) __bf16 bf16x16;
typedef __attribute__((ext_vector_type(8)))  __bf16 bf16x8;
typedef __attribute__((ext_vector_type(4)))  __bf16 bf16x4;
typedef __attribute__((ext_vector_type(8)))  float  f32x8;
typedef __attribute__((ext_vector_type(4)))  unsigned int u32x4;
typedef __attribute__((ext_vector_type(8)))  int    i32x8;
typedef __attribute__((ext_vector_type(4)))  int    i32x4;

#define MTOK   73728        // 8 * 96 * 96 tokens
#define NTOK   9216         // tokens per frame
#define DMODEL 256
#define HEADS  8
#define DH     32

#if defined(__HIP_DEVICE_COMPILE__) && __has_builtin(__builtin_amdgcn_tensor_load_to_lds)
#define USE_TDM 1
#else
#define USE_TDM 0
#endif

static __device__ __forceinline__ bf16x16 cat8(bf16x8 lo, bf16x8 hi) {
    return __builtin_shufflevector(lo, hi, 0,1,2,3,4,5,6,7,8,9,10,11,12,13,14,15);
}

#if USE_TDM
// DMA a 64-row x 32-col f32 tile (row stride 256 floats) into LDS,
// with 16B of LDS padding after each 128B row (rows land at 144B stride).
static __device__ __forceinline__ void tdm_load_b_tile(const float* gsrc,
                                                       unsigned lds_off) {
    unsigned long long ga = (unsigned long long)(uintptr_t)gsrc;
    u32x4 g0;
    g0[0] = 1u;                                             // count=1, user D#
    g0[1] = lds_off;                                        // lds_addr (bytes)
    g0[2] = (unsigned)(ga & 0xFFFFFFFFu);                   // global_addr[31:0]
    g0[3] = (unsigned)((ga >> 32) & 0x01FFFFFFu) | (2u << 30); // [56:32] | type=2
    i32x8 g1;
    g1[0] = (int)((2u << 16)        // data_size = 4B
                | (1u << 20)        // pad_enable
                | (4u << 22)        // pad_interval: every 32 DWORDs (128B)
                | (3u << 25));      // pad_amount: 4 DWORDs (16B)
    g1[1] = (int)(32u << 16);       // tensor_dim0 = 32 (low 16 @ bits 63:48)
    g1[2] = (int)(64u << 16);       // tensor_dim1 = 64 (low 16 @ bits 95:80)
    g1[3] = (int)(32u << 16);       // tile_dim0 = 32 (bits 127:112)
    g1[4] = 64;                     // tile_dim1 = 64 (bits 143:128)
    g1[5] = 256;                    // tensor_dim0_stride = 256 elements
    g1[6] = 0;
    g1[7] = 0;
    i32x4 z4 = (i32x4)0;
#if __clang_major__ >= 23
    i32x8 z8 = (i32x8)0;
    __builtin_amdgcn_tensor_load_to_lds(g0, g1, z4, z4, z8, 0);
#else
    __builtin_amdgcn_tensor_load_to_lds(g0, g1, z4, z4, 0);
#endif
}
#endif

// ---------------------------------------------------------------------------
// Generic GEMM:  C[M,N] = A[M,256] @ W[N,256]^T (+bias) (+f32 residual)
// Block tile 128x64, K chunks of 32, 8 waves in 4(M)x2(N) grid,
// each wave: 2x2 subtiles of 16x16 -> 4 v_wmma_f32_16x16x32_bf16 per chunk.
// B (weights) tile is TDM-DMA'd into LDS as f32 with HW padding; converted to
// bf16 at fragment build. A tile staged through VGPRs (f32->bf16 for K1).
// ---------------------------------------------------------------------------
template<bool A_F32, bool BIAS, bool RES_F32OUT>
__global__ __launch_bounds__(256)
void gemm_wmma_bf16(const void* __restrict__ Ain, const float* __restrict__ W,
                    const float* __restrict__ bias, const float* __restrict__ residual,
                    void* __restrict__ Cout, int N) {
    constexpr int BM = 128, LDA = 40;      // bf16 A tile, +8 pad
    constexpr int LDBF = 36;               // f32 B tile row stride (32 + 4 pad)
    __shared__ __bf16 sA[BM * LDA];        // 10240 B
    __shared__ float  sBf[64 * LDBF];      // 9216 B

    const int tid  = threadIdx.x;
    const int lane = tid & 31;
    const int wid  = tid >> 5;
    const int wm   = wid & 3;          // 4 waves along M (32 rows each)
    const int wn   = wid >> 2;         // 2 waves along N (32 cols each)
    const int lrow = lane & 15;
    const int half = lane >> 4;
    const int bm   = blockIdx.x * BM;
    const int bn   = blockIdx.y * 64;

    f32x8 acc[2][2] = {};

    for (int kb = 0; kb < DMODEL; kb += 32) {
        // ---- B tile (64 rows of W x 32 k, f32): TDM DMA into padded LDS ----
#if USE_TDM
        if (wid == 0)
            tdm_load_b_tile(W + (size_t)bn * DMODEL + kb,
                            (unsigned)(uintptr_t)(void*)sBf);
#else
        #pragma unroll
        for (int i = 0; i < 2; ++i) {
            int idx = tid + i * 256;                 // 512 float4 chunks
            int r = idx >> 3, k4 = idx & 7;
            float4 v = *(const float4*)(W + (size_t)(bn + r) * DMODEL + kb + k4 * 4);
            *(float4*)&sBf[r * LDBF + k4 * 4] = v;
        }
#endif
        // ---- stage A tile (128 x 32) into LDS as bf16 ----
        if (A_F32) {
            const float* A = (const float*)Ain;
            #pragma unroll
            for (int i = 0; i < 4; ++i) {
                int idx = tid + i * 256;             // 1024 float4 chunks
                int r = idx >> 3, k4 = idx & 7;
                const float* src = A + (size_t)(bm + r) * DMODEL + kb + k4 * 4;
                if (kb + 32 < DMODEL) __builtin_prefetch(src + 32, 0, 1);
                float4 v = *(const float4*)src;
                bf16x4 b; b.x = (__bf16)v.x; b.y = (__bf16)v.y;
                          b.z = (__bf16)v.z; b.w = (__bf16)v.w;
                *(bf16x4*)&sA[r * LDA + k4 * 4] = b;
            }
        } else {
            const __bf16* A = (const __bf16*)Ain;
            #pragma unroll
            for (int i = 0; i < 2; ++i) {
                int idx = tid + i * 256;             // 512 16-byte chunks
                int r = idx >> 2, kc = idx & 3;
                const __bf16* src = A + (size_t)(bm + r) * DMODEL + kb + kc * 8;
                if (kb + 32 < DMODEL) __builtin_prefetch(src + 32, 0, 1);
                *(bf16x8*)&sA[r * LDA + kc * 8] = *(const bf16x8*)src;
            }
        }
#if USE_TDM
        if (wid == 0) __builtin_amdgcn_s_wait_tensorcnt(0);
#endif
        __syncthreads();

        // ---- fragments + WMMA ----
        bf16x16 af[2], bfr[2];
        #pragma unroll
        for (int mi = 0; mi < 2; ++mi) {
            int row = wm * 32 + mi * 16 + lrow;
            bf16x8 lo = *(const bf16x8*)&sA[row * LDA + 16 * half];
            bf16x8 hi = *(const bf16x8*)&sA[row * LDA + 16 * half + 8];
            af[mi] = cat8(lo, hi);                   // k = 16*half + j
        }
        #pragma unroll
        for (int ni = 0; ni < 2; ++ni) {
            int col = wn * 32 + ni * 16 + lrow;
            const float* bp = &sBf[col * LDBF + 16 * half];   // k = 16*half + j
            f32x8 lo = *(const f32x8*)bp;
            f32x8 hi = *(const f32x8*)(bp + 8);
            bf16x16 b;
            #pragma unroll
            for (int j = 0; j < 8; ++j) {
                b[j]     = (__bf16)lo[j];
                b[8 + j] = (__bf16)hi[j];
            }
            bfr[ni] = b;
        }
        #pragma unroll
        for (int mi = 0; mi < 2; ++mi)
            #pragma unroll
            for (int ni = 0; ni < 2; ++ni)
                acc[mi][ni] = __builtin_amdgcn_wmma_f32_16x16x32_bf16(
                    false, af[mi], false, bfr[ni], (short)0, acc[mi][ni], false, false);
        __syncthreads();
    }

    // ---- epilogue: bias, residual, store ----
    #pragma unroll
    for (int ni = 0; ni < 2; ++ni) {
        int col = bn + wn * 32 + ni * 16 + lrow;
        float bv = BIAS ? bias[col] : 0.0f;
        #pragma unroll
        for (int mi = 0; mi < 2; ++mi) {
            #pragma unroll
            for (int r = 0; r < 8; ++r) {
                int row = bm + wm * 32 + mi * 16 + r + 8 * half;  // D: m = r + 8*half
                float v = acc[mi][ni][r] + bv;
                if (RES_F32OUT) {
                    ((float*)Cout)[(size_t)row * N + col] =
                        v + residual[(size_t)row * DMODEL + col];
                } else {
                    ((__bf16*)Cout)[(size_t)row * N + col] = (__bf16)v;
                }
            }
        }
    }
}

// ---------------------------------------------------------------------------
// K2: per (frame, head): spatial softmax of K over 9216 tokens, then
//     ctx[d][e] = sum_n softmax_k[n][d] * v[n][e]; stored TRANSPOSED:
//     ctxT[f][h][e][d]  (so K3 B-fragments read contiguous k runs)
// ---------------------------------------------------------------------------
__global__ __launch_bounds__(256)
void ctx_kernel(const __bf16* __restrict__ qkv, float* __restrict__ ctxT) {
    const int f = blockIdx.x >> 3, h = blockIdx.x & 7;
    const int tid = threadIdx.x, c = tid & 31, g = tid >> 5;
    const size_t base = (size_t)f * NTOK * 768;
    const int kcol = 256 + h * DH, vcol = 512 + h * DH;

    __shared__ float red[8][32];
    __shared__ float mx[32], rs[32];
    __shared__ float kn[8][32];

    // pass 1: per-channel max over tokens
    float lm = -1e30f;
    for (int t = g; t < NTOK; t += 8)
        lm = fmaxf(lm, (float)qkv[base + (size_t)t * 768 + kcol + c]);
    red[g][c] = lm; __syncthreads();
    if (g == 0) { float m = red[0][c];
                  for (int i = 1; i < 8; ++i) m = fmaxf(m, red[i][c]); mx[c] = m; }
    __syncthreads();
    const float m = mx[c];

    // pass 2: per-channel sum of exp
    float ls = 0.0f;
    for (int t = g; t < NTOK; t += 8)
        ls += __expf((float)qkv[base + (size_t)t * 768 + kcol + c] - m);
    red[g][c] = ls; __syncthreads();
    if (g == 0) { float s = 0.0f;
                  for (int i = 0; i < 8; ++i) s += red[i][c]; rs[c] = 1.0f / s; }
    __syncthreads();
    const float rinv = rs[c];

    // pass 3: accumulate ctx; thread (e=c, group g) owns column e partials
    float acc[32];
    #pragma unroll
    for (int d = 0; d < 32; ++d) acc[d] = 0.0f;
    for (int t = g; t < NTOK; t += 8) {
        kn[g][c] = __expf((float)qkv[base + (size_t)t * 768 + kcol + c] - m) * rinv;
        __syncthreads();
        float v = (float)qkv[base + (size_t)t * 768 + vcol + c];
        #pragma unroll
        for (int d = 0; d < 32; ++d) acc[d] += kn[g][d] * v;
        __syncthreads();
    }
    // reduce over the 8 token groups, write ctxT[e][d]
    for (int d = 0; d < 32; ++d) {
        red[g][c] = acc[d]; __syncthreads();
        if (g == 0) {
            float s = 0.0f;
            for (int i = 0; i < 8; ++i) s += red[i][c];
            ctxT[((((size_t)f * 8 + h) * 32) + c) * 32 + d] = s;
        }
        __syncthreads();
    }
}

// ---------------------------------------------------------------------------
// K3: channel softmax of Q per token, then out = Qs @ ctx per head via WMMA
//     (M=16 tokens per wave, N=32 per head in two 16-wide tiles, K=32 exact).
//     attn is head-major merged: col = h*32 + e  (bf16)
// ---------------------------------------------------------------------------
__global__ __launch_bounds__(256)
void attnout_kernel(const __bf16* __restrict__ qkv, const float* __restrict__ ctxT,
                    __bf16* __restrict__ attn) {
    const float scale = 0.17677669529663687f;     // 32^-0.5
    const int tid = threadIdx.x, lane = tid & 31, wid = tid >> 5;
    const int f = blockIdx.x / 72, ttile = blockIdx.x % 72;
    const int tok0 = f * NTOK + ttile * 128;

    __shared__ __bf16 sctx[8 * 32 * 32];   // [h][e][d]
    __shared__ __bf16 sq [8][16][32];      // raw q, per wave
    __shared__ __bf16 sqs[8][16][32];      // softmaxed q, per wave

    { // stage all 8 heads' ctxT for this frame as bf16
        const float* src = ctxT + (size_t)f * 8 * 32 * 32;
        for (int i = tid; i < 8 * 32 * 32; i += 256) sctx[i] = (__bf16)src[i];
    }
    __syncthreads();

    for (int h = 0; h < HEADS; ++h) {
        // stage 16 tokens x 32 ch of raw q for this wave
        {
            int tk = lane >> 1;
            const __bf16* qr = qkv + (size_t)(tok0 + wid * 16 + tk) * 768
                                   + h * DH + (lane & 1) * 16;
            *(bf16x8*)&sq[wid][tk][(lane & 1) * 16]     = *(const bf16x8*)qr;
            *(bf16x8*)&sq[wid][tk][(lane & 1) * 16 + 8] = *(const bf16x8*)(qr + 8);
        }
        __syncthreads();
        // channel softmax (each lane pair handles one token; halves of channels)
        {
            int t = lane & 15, ch = lane >> 4;
            float vals[32]; float vmax = -1e30f;
            #pragma unroll
            for (int cc = 0; cc < 32; ++cc) {
                vals[cc] = (float)sq[wid][t][cc]; vmax = fmaxf(vmax, vals[cc]);
            }
            float s = 0.0f;
            #pragma unroll
            for (int cc = 0; cc < 32; ++cc) { vals[cc] = __expf(vals[cc] - vmax); s += vals[cc]; }
            float r = scale / s;
            #pragma unroll
            for (int cc = 0; cc < 16; ++cc)
                sqs[wid][t][ch * 16 + cc] = (__bf16)(vals[ch * 16 + cc] * r);
        }
        __syncthreads();
        // WMMA: A = Qs (16x32), B = ctx (32x16 per tile)
        const int m = lane & 15, half = lane >> 4;
        bf16x16 a = cat8(*(const bf16x8*)&sqs[wid][m][16 * half],
                         *(const bf16x8*)&sqs[wid][m][16 * half + 8]);
        f32x8 acc0 = {}, acc1 = {};
        #pragma unroll
        for (int ni = 0; ni < 2; ++ni) {
            int n = ni * 16 + m;
            const __bf16* bp = &sctx[(h * 32 + n) * 32 + 16 * half];
            bf16x16 b = cat8(*(const bf16x8*)bp, *(const bf16x8*)(bp + 8));
            f32x8& acc = ni ? acc1 : acc0;
            acc = __builtin_amdgcn_wmma_f32_16x16x32_bf16(
                      false, a, false, b, (short)0, acc, false, false);
        }
        #pragma unroll
        for (int ni = 0; ni < 2; ++ni) {
            f32x8 acc = ni ? acc1 : acc0;
            int col = h * DH + ni * 16 + m;
            #pragma unroll
            for (int r = 0; r < 8; ++r) {
                int row = tok0 + wid * 16 + r + 8 * half;
                attn[(size_t)row * DMODEL + col] = (__bf16)acc[r];
            }
        }
        __syncthreads();
    }
}

// ---------------------------------------------------------------------------
// launch
// ---------------------------------------------------------------------------
extern "C" void kernel_launch(void* const* d_in, const int* in_sizes, int n_in,
                              void* d_out, int out_size, void* d_ws, size_t ws_size,
                              hipStream_t stream) {
    const float* x     = (const float*)d_in[0];   // [73728, 256]
    const float* w_qkv = (const float*)d_in[1];   // [768, 256]
    const float* w_out = (const float*)d_in[2];   // [256, 256]
    const float* b_out = (const float*)d_in[3];   // [256]
    const float* w_lin = (const float*)d_in[4];   // [256, 256]
    const float* b_lin = (const float*)d_in[5];   // [256]
    float* out = (float*)d_out;

    // workspace layout (bytes): qkv bf16 | ctxT f32 | attn bf16 | y1 bf16
    char* ws = (char*)d_ws;
    __bf16* qkv  = (__bf16*)(ws);                                   // 113,246,208
    float*  ctxT = (float*) (ws + 113246208);                       //     262,144
    __bf16* attn = (__bf16*)(ws + 113246208 + 262144);              //  37,748,736
    __bf16* y1   = (__bf16*)(ws + 113246208 + 262144 + 37748736);   //  37,748,736

    // K1: qkv = x @ w_qkv^T         (bf16 out, no bias)
    gemm_wmma_bf16<true, false, false>
        <<<dim3(MTOK / 128, 768 / 64), 256, 0, stream>>>(x, w_qkv, nullptr, nullptr, qkv, 768);

    // K2: spatial softmax(K) + ctx per (frame, head)
    ctx_kernel<<<64, 256, 0, stream>>>(qkv, ctxT);

    // K3: channel softmax(Q), out = Qs @ ctx, head-major merge
    attnout_kernel<<<576, 256, 0, stream>>>(qkv, ctxT, attn);

    // K4a: y1 = attn @ w_out^T + b_out  (bf16 out)
    gemm_wmma_bf16<false, true, false>
        <<<dim3(MTOK / 128, 256 / 64), 256, 0, stream>>>(attn, w_out, b_out, nullptr, y1, 256);

    // K4b: out = y1 @ w_lin^T + b_lin + x  (f32 out)
    gemm_wmma_bf16<false, true, true>
        <<<dim3(MTOK / 128, 256 / 64), 256, 0, stream>>>(y1, w_lin, b_lin, x, out, 256);
}
